// TriangleMultiplicativeUpdate_84937273245983
// MI455X (gfx1250) — compile-verified, matrix operand order checked by reading, not run on hardware
//
#include <hip/hip_runtime.h>
#include <hip/hip_bf16.h>

typedef __attribute__((ext_vector_type(16))) __bf16 bf16x16;
typedef __attribute__((ext_vector_type(8)))  float  v8f;

union FragB { bf16x16 v; uint4 u[2]; };

__device__ __forceinline__ __bf16 f2bf(float f) {
  unsigned u = __builtin_bit_cast(unsigned, f);
  u += 0x7FFFu + ((u >> 16) & 1u);           // round-to-nearest-even
  unsigned short s = (unsigned short)(u >> 16);
  return __builtin_bit_cast(__bf16, s);
}
// pack two floats into packed bf16x2 (lo in [15:0], hi in [31:16])
__device__ __forceinline__ unsigned pack_bf16x2(float a, float b) {
  unsigned ua = __builtin_bit_cast(unsigned, a);
  ua += 0x7FFFu + ((ua >> 16) & 1u);
  unsigned ub = __builtin_bit_cast(unsigned, b);
  ub += 0x7FFFu + ((ub >> 16) & 1u);
  return (ua >> 16) | (ub & 0xFFFF0000u);
}
__device__ __forceinline__ float bf2f(__bf16 b) {
  unsigned u = ((unsigned)__builtin_bit_cast(unsigned short, b)) << 16;
  return __builtin_bit_cast(float, u);
}
// fast sigmoid: v_exp_f32 + v_rcp_f32 (no IEEE div expansion)
__device__ __forceinline__ float sigmoidf_(float x) {
  return __builtin_amdgcn_rcpf(1.0f + __expf(-x));
}
__device__ __forceinline__ v8f zero8() {
  v8f v;
#pragma unroll
  for (int r = 0; r < 8; ++r) v[r] = 0.f;
  return v;
}

// ---------------------------------------------------------------------------
// k0: convert weights to bf16. wcat = [w_ab_g(256x128); w_ab_p(256x128); w_g(128x128)]
// ---------------------------------------------------------------------------
__global__ __launch_bounds__(256) void k0_cvt(
    const float* __restrict__ w_ab_p, const float* __restrict__ w_ab_g,
    const float* __restrict__ w_g, const float* __restrict__ w_z,
    __bf16* __restrict__ wcat, __bf16* __restrict__ wz_bf)
{
  int t = blockIdx.x * 256 + threadIdx.x;
  if      (t < 32768) wcat[t] = f2bf(w_ab_g[t]);
  else if (t < 65536) wcat[t] = f2bf(w_ab_p[t - 32768]);
  else if (t < 81920) wcat[t] = f2bf(w_g[t - 65536]);
  else if (t < 98304) wz_bf[t - 81920] = f2bf(w_z[t - 81920]);
}

// ---------------------------------------------------------------------------
// k1: fused LayerNorm-in + (gate,proj,g) projections via bf16 WMMA.
// One block = 16 consecutive (i,j) rows (same i, 16 consecutive j).
// Writes a_t/b_t transposed to [h][i][j] (bf16) and g_ws [row][d] (bf16).
// ---------------------------------------------------------------------------
__global__ __launch_bounds__(256) void k1_ln_proj(
    const float* __restrict__ z, const float* __restrict__ mask,
    const __bf16* __restrict__ wcat,
    const float* __restrict__ b_ab_p, const float* __restrict__ b_ab_g,
    const float* __restrict__ b_g,
    const float* __restrict__ ln_g, const float* __restrict__ ln_b,
    __bf16* __restrict__ a_t, __bf16* __restrict__ b_t,
    __bf16* __restrict__ g_ws)
{
  __shared__ float s_z[16 * 132];
  __shared__ float s_red[16 * 32];
  __shared__ float s_stat[16 * 2];
  __shared__ __align__(16) __bf16 s_znb[16 * 136];
  __shared__ __align__(16) __bf16 s_ab[256 * 16];

  const int tid  = threadIdx.x;
  const int lane = tid & 31;
  const int wave = tid >> 5;
  const int row0 = blockIdx.x * 16;
  const int i_idx = row0 >> 9;
  const int j0    = row0 & 511;

  // ---- load z tile + LayerNorm partials ----
  {
    int r = tid >> 4, grp = tid & 15;
    const float* zp = z + (size_t)(row0 + r) * 128 + grp * 8;
    float4 v0 = *(const float4*)zp;
    float4 v1 = *(const float4*)(zp + 4);
    float* dst = &s_z[r * 132 + grp * 8];
    *(float4*)dst = v0; *(float4*)(dst + 4) = v1;
    float s1 = v0.x + v0.y + v0.z + v0.w + v1.x + v1.y + v1.z + v1.w;
    float s2 = v0.x*v0.x + v0.y*v0.y + v0.z*v0.z + v0.w*v0.w
             + v1.x*v1.x + v1.y*v1.y + v1.z*v1.z + v1.w*v1.w;
    s_red[(r * 16 + grp) * 2 + 0] = s1;
    s_red[(r * 16 + grp) * 2 + 1] = s2;
  }
  __syncthreads();
  if (tid < 16) {
    float s1 = 0.f, s2 = 0.f;
    for (int g2 = 0; g2 < 16; ++g2) {
      s1 += s_red[(tid * 16 + g2) * 2 + 0];
      s2 += s_red[(tid * 16 + g2) * 2 + 1];
    }
    float mu  = s1 * (1.f / 128.f);
    float var = s2 * (1.f / 128.f) - mu * mu;
    s_stat[tid * 2 + 0] = mu;
    s_stat[tid * 2 + 1] = rsqrtf(var + 1e-5f);
  }
  __syncthreads();
  {
    int r = tid >> 4, grp = tid & 15;
    float mu = s_stat[r * 2], inv = s_stat[r * 2 + 1];
    int c0 = grp * 8;
    float zn[8];
#pragma unroll
    for (int q = 0; q < 8; ++q) {
      int c = c0 + q;
      zn[q] = (s_z[r * 132 + c] - mu) * inv * ln_g[c] + ln_b[c];
    }
    uint4 pk;
    pk.x = pack_bf16x2(zn[0], zn[1]);
    pk.y = pack_bf16x2(zn[2], zn[3]);
    pk.z = pack_bf16x2(zn[4], zn[5]);
    pk.w = pack_bf16x2(zn[6], zn[7]);
    *(uint4*)&s_znb[r * 136 + c0] = pk;   // one ds_store_b128
  }
  __syncthreads();

  // ---- WMMA projection: 24 jobs (16 gate+proj pairs, 8 g-tiles), 3 per wave
  const int rr  = lane & 15;
  const int kh  = (lane >> 4) << 3;   // K half-offset for A/B fragments
  const int mhi = (lane >> 4) << 3;   // M half-offset for C/D fragments

  for (int q = 0; q < 3; ++q) {
    int job = wave * 3 + q;           // wave-uniform
    if (job < 16) {
      v8f accG = zero8(), accP = zero8();
#pragma unroll
      for (int kc = 0; kc < 4; ++kc) {
        int k0 = kc * 32;
        FragB A, Bg, Bp;
        const __bf16* ap = &s_znb[rr * 136 + k0 + kh];
        A.u[0] = *(const uint4*)ap; A.u[1] = *(const uint4*)(ap + 16);
        const __bf16* gp = wcat + (size_t)(job * 16 + rr) * 128 + k0 + kh;
        Bg.u[0] = *(const uint4*)gp; Bg.u[1] = *(const uint4*)(gp + 16);
        const __bf16* pp = wcat + (size_t)(256 + job * 16 + rr) * 128 + k0 + kh;
        Bp.u[0] = *(const uint4*)pp; Bp.u[1] = *(const uint4*)(pp + 16);
        accG = __builtin_amdgcn_wmma_f32_16x16x32_bf16(false, A.v, false, Bg.v, (short)0, accG, false, false);
        accP = __builtin_amdgcn_wmma_f32_16x16x32_bf16(false, A.v, false, Bp.v, (short)0, accP, false, false);
      }
      int nn = job * 16 + rr;
      float bg = b_ab_g[nn], bp = b_ab_p[nn];
#pragma unroll
      for (int r = 0; r < 8; r += 2) {     // pack adjacent rows m, m+1
        int m = r + mhi;
        float mk0 = mask[row0 + m];
        float mk1 = mask[row0 + m + 1];
        float v0 = sigmoidf_(accG[r]     + bg) * mk0 * (accP[r]     + bp);
        float v1 = sigmoidf_(accG[r + 1] + bg) * mk1 * (accP[r + 1] + bp);
        *(unsigned*)&s_ab[nn * 16 + m] = pack_bf16x2(v0, v1);
      }
    } else {
      int gi = job - 16;
      v8f acc = zero8();
#pragma unroll
      for (int kc = 0; kc < 4; ++kc) {
        int k0 = kc * 32;
        FragB A, B;
        const __bf16* ap = &s_znb[rr * 136 + k0 + kh];
        A.u[0] = *(const uint4*)ap; A.u[1] = *(const uint4*)(ap + 16);
        const __bf16* wp = wcat + (size_t)(512 + gi * 16 + rr) * 128 + k0 + kh;
        B.u[0] = *(const uint4*)wp; B.u[1] = *(const uint4*)(wp + 16);
        acc = __builtin_amdgcn_wmma_f32_16x16x32_bf16(false, A.v, false, B.v, (short)0, acc, false, false);
      }
      int d = gi * 16 + rr;
      float bgc = b_g[d];
#pragma unroll
      for (int r = 0; r < 8; ++r) {
        int m = r + mhi;
        g_ws[(size_t)(row0 + m) * 128 + d] = f2bf(sigmoidf_(acc[r] + bgc));
      }
    }
  }
  __syncthreads();

  // ---- transposed writeout: a_t/b_t are [h][i][j], j contiguous
  {
    int n = tid;  // 0..255 = output channel within ab
    const uint4* src = (const uint4*)&s_ab[n * 16];
    uint4 lo = src[0], hi = src[1];
    __bf16* dst;
    if (n < 128) dst = a_t + ((size_t)n * 512 + i_idx) * 512 + j0;
    else         dst = b_t + ((size_t)(n - 128) * 512 + i_idx) * 512 + j0;
    *(uint4*)dst = lo; *(uint4*)(dst + 8) = hi;
  }
}

// ---------------------------------------------------------------------------
// k2: triangle GEMM (128 batched 512x512x512 NT-GEMMs in bf16 WMMA),
// fused LN-over-H from accumulators, fused output projection + gate.
// One block = 16x16 (i,j) tile, all 128 h; wave w owns h in [16w,16w+16).
// ---------------------------------------------------------------------------
__global__ __launch_bounds__(256) void k2_tri(
    const __bf16* __restrict__ a_t, const __bf16* __restrict__ b_t,
    const __bf16* __restrict__ wz_bf, const float* __restrict__ b_z,
    const float* __restrict__ ln_g, const float* __restrict__ ln_b,
    const __bf16* __restrict__ g_ws, float* __restrict__ out)
{
  __shared__ float s_red[256 * 16];                 // [pos][wave*2 + {s1,s2}]
  __shared__ __align__(16) __bf16 s_xn[256 * 136];  // normalized x, [pos][h]

  const int tid  = threadIdx.x;
  const int lane = tid & 31;
  const int wave = tid >> 5;
  const int it = blockIdx.x & 31, jt = blockIdx.x >> 5;
  const int i0 = it * 16, j0 = jt * 16;
  const int rr  = lane & 15;
  const int kh  = (lane >> 4) << 3;
  const int mhi = (lane >> 4) << 3;

  v8f acc[16];
#pragma unroll
  for (int f = 0; f < 16; ++f) acc[f] = zero8();

  const __bf16* a_base = a_t + ((size_t)wave * 16 * 512 + i0 + rr) * 512 + kh;
  const __bf16* b_base = b_t + ((size_t)wave * 16 * 512 + j0 + rr) * 512 + kh;

  // ---- main K loop: x_h[16x16] += A_h(16x32) * B_h(32x16)^T, 16 h per wave
  for (int kc = 0; kc < 16; ++kc) {
    int k0 = kc * 32;
    if (kc < 15) {  // prefetch next k-chunk (L2 -> WGP)
      __builtin_prefetch(a_base + k0 + 32, 0, 1);
      __builtin_prefetch(b_base + k0 + 32, 0, 1);
    }
#pragma unroll
    for (int f = 0; f < 16; ++f) {
      FragB A, B;
      const __bf16* ap = a_base + (size_t)f * 512 * 512 + k0;
      A.u[0] = *(const uint4*)ap; A.u[1] = *(const uint4*)(ap + 16);
      const __bf16* bp = b_base + (size_t)f * 512 * 512 + k0;
      B.u[0] = *(const uint4*)bp; B.u[1] = *(const uint4*)(bp + 16);
      acc[f] = __builtin_amdgcn_wmma_f32_16x16x32_bf16(false, A.v, false, B.v, (short)0, acc[f], false, false);
    }
  }

  // ---- LN over h: per-lane partials from accumulators, cross-wave via LDS
  float s1[8], s2[8];
#pragma unroll
  for (int r = 0; r < 8; ++r) { s1[r] = 0.f; s2[r] = 0.f; }
#pragma unroll
  for (int f = 0; f < 16; ++f)
#pragma unroll
    for (int r = 0; r < 8; ++r) { float v = acc[f][r]; s1[r] += v; s2[r] += v * v; }
#pragma unroll
  for (int r = 0; r < 8; ++r) {
    int pos = (r + mhi) * 16 + rr;
    s_red[pos * 16 + wave * 2 + 0] = s1[r];
    s_red[pos * 16 + wave * 2 + 1] = s2[r];
  }
  __syncthreads();
  float mu[8], inv[8];
#pragma unroll
  for (int r = 0; r < 8; ++r) {
    int pos = (r + mhi) * 16 + rr;
    float t1 = 0.f, t2 = 0.f;
#pragma unroll
    for (int w2 = 0; w2 < 8; ++w2) {
      t1 += s_red[pos * 16 + w2 * 2 + 0];
      t2 += s_red[pos * 16 + w2 * 2 + 1];
    }
    mu[r]  = t1 * (1.f / 128.f);
    inv[r] = rsqrtf(t2 * (1.f / 128.f) - mu[r] * mu[r] + 1e-5f);
  }
#pragma unroll
  for (int f = 0; f < 16; f += 2) {       // pack adjacent channels h, h+1
    int h = wave * 16 + f;
    float gam0 = ln_g[h],     bet0 = ln_b[h];
    float gam1 = ln_g[h + 1], bet1 = ln_b[h + 1];
#pragma unroll
    for (int r = 0; r < 8; ++r) {
      int pos = (r + mhi) * 16 + rr;
      float v0 = (acc[f][r]     - mu[r]) * inv[r] * gam0 + bet0;
      float v1 = (acc[f + 1][r] - mu[r]) * inv[r] * gam1 + bet1;
      *(unsigned*)&s_xn[pos * 136 + h] = pack_bf16x2(v0, v1);
    }
  }
  __syncthreads();

  // ---- output projection: (256 pos x 128 h) * w_z^T (128x128) + b_z, * gate
  for (int q = 0; q < 16; ++q) {
    int job = wave * 16 + q;          // wave-uniform
    int Mt = job >> 3, Nt = job & 7;
    v8f a2 = zero8();
#pragma unroll
    for (int kc = 0; kc < 4; ++kc) {
      int k0 = kc * 32;
      FragB A, B;
      const __bf16* ap = &s_xn[(Mt * 16 + rr) * 136 + k0 + kh];
      A.u[0] = *(const uint4*)ap; A.u[1] = *(const uint4*)(ap + 16);
      const __bf16* bp = wz_bf + (size_t)(Nt * 16 + rr) * 128 + k0 + kh;
      B.u[0] = *(const uint4*)bp; B.u[1] = *(const uint4*)(bp + 16);
      a2 = __builtin_amdgcn_wmma_f32_16x16x32_bf16(false, A.v, false, B.v, (short)0, a2, false, false);
    }
    int c = Nt * 16 + rr;
    float bz = b_z[c];
#pragma unroll
    for (int r = 0; r < 8; ++r) {
      int m  = Mt * 16 + r + mhi;     // pos = i_local*16 + j_local
      int il = m >> 4, jl = m & 15;
      size_t rowg = (size_t)(i0 + il) * 512 + (j0 + jl);
      float gv = bf2f(g_ws[rowg * 128 + c]);
      out[rowg * 128 + c] = (a2[r] + bz) * gv;
    }
  }
}

// ---------------------------------------------------------------------------
extern "C" void kernel_launch(void* const* d_in, const int* in_sizes, int n_in,
                              void* d_out, int out_size, void* d_ws, size_t ws_size,
                              hipStream_t stream) {
  (void)in_sizes; (void)n_in; (void)out_size; (void)ws_size;
  const float* z       = (const float*)d_in[0];
  const float* mask    = (const float*)d_in[1];
  const float* w_ab_p  = (const float*)d_in[2];
  const float* b_ab_p  = (const float*)d_in[3];
  const float* w_ab_g  = (const float*)d_in[4];
  const float* b_ab_g  = (const float*)d_in[5];
  const float* w_g     = (const float*)d_in[6];
  const float* b_g     = (const float*)d_in[7];
  const float* w_z     = (const float*)d_in[8];
  const float* b_z     = (const float*)d_in[9];
  const float* ln_in_g = (const float*)d_in[10];
  const float* ln_in_b = (const float*)d_in[11];
  const float* ln_out_g= (const float*)d_in[12];
  const float* ln_out_b= (const float*)d_in[13];
  float* out = (float*)d_out;

  char* ws = (char*)d_ws;
  size_t off = 0;
  __bf16* wcat  = (__bf16*)(ws + off); off += (size_t)640 * 128 * 2;  // 163840
  __bf16* wz_bf = (__bf16*)(ws + off); off += (size_t)128 * 128 * 2;  // 32768
  off = (off + 255) & ~(size_t)255;
  __bf16* a_t  = (__bf16*)(ws + off); off += (size_t)128 * 512 * 512 * 2;
  __bf16* b_t  = (__bf16*)(ws + off); off += (size_t)128 * 512 * 512 * 2;
  __bf16* g_ws = (__bf16*)(ws + off); off += (size_t)512 * 512 * 128 * 2;

  hipLaunchKernelGGL(k0_cvt, dim3(384), dim3(256), 0, stream,
                     w_ab_p, w_ab_g, w_g, w_z, wcat, wz_bf);
  hipLaunchKernelGGL(k1_ln_proj, dim3(16384), dim3(256), 0, stream,
                     z, mask, wcat, b_ab_p, b_ab_g, b_g, ln_in_g, ln_in_b,
                     a_t, b_t, g_ws);
  hipLaunchKernelGGL(k2_tri, dim3(1024), dim3(256), 0, stream,
                     a_t, b_t, wz_bf, b_z, ln_out_g, ln_out_b, g_ws, out);
}
